// QLSTMGen313_65481071410163
// MI455X (gfx1250) — compile-verified, matrix operand order checked by reading, not run on hardware
//
#include <hip/hip_runtime.h>
#include <stdint.h>

typedef __bf16 bf16;
typedef __attribute__((ext_vector_type(16))) __bf16 v16bf;
typedef __attribute__((ext_vector_type(8)))  float  v8f;

#define T_STEPS 1024
#define BATCH   64
#define DIN     512
#define HID     512
#define G4      2048            // 4*HID
#define TB      (T_STEPS*BATCH) // 65536
#define NBLK    16              // persistent scan blocks
#define SLICE   32              // hidden columns owned per scan block
#define LDH     520             // padded LDS row stride in bf16 (512+8 -> 4-bank skew)

union FragBF { v16bf v; uint4 u[2]; };
union Pack8  { bf16 h[8]; uint4 u; };

// Load a 16x32 bf16 WMMA operand fragment from a row-major [rows][ld] matrix.
// lane r (0..15) carries row r; half=0 -> K {0..7,16..23}, half=1 -> K {8..15,24..31}.
__device__ __forceinline__ v16bf load_frag(const bf16* __restrict__ base,
                                           int row0, int k0, int ld, int r, int half) {
  const bf16* p = base + (size_t)(row0 + r) * ld + k0 + half * 8;
  FragBF f;
  f.u[0] = *(const uint4*)(p);
  f.u[1] = *(const uint4*)(p + 16);
  return f.v;
}

__device__ __forceinline__ float sigm_f(float x) { return 1.0f / (1.0f + __expf(-x)); }
__device__ __forceinline__ float tanh_f(float x) {
  x = fminf(fmaxf(x, -15.0f), 15.0f);
  float e = __expf(2.0f * x);
  return (e - 1.0f) / (e + 1.0f);
}

// ---------------- init: zero barrier counter + h double buffers ----------------
__global__ void k_init(unsigned* __restrict__ cnt, bf16* __restrict__ h0, bf16* __restrict__ h1) {
  int i = blockIdx.x * 256 + threadIdx.x;
  if (i == 0) *cnt = 0u;
  if (i < BATCH * HID) { h0[i] = (bf16)0.0f; h1[i] = (bf16)0.0f; }
}

// ---------------- convert gate weights: split x/h halves, fp32 -> bf16 ----------------
__global__ void k_cvtW(const float* __restrict__ Wf, const float* __restrict__ Wi,
                       const float* __restrict__ Wg, const float* __restrict__ Wo,
                       bf16* __restrict__ Wx, bf16* __restrict__ Wh) {
  int idx = blockIdx.x * 256 + threadIdx.x;       // over 2048*512
  if (idx >= G4 * 512) return;
  int row = idx >> 9;                             // gate row 0..2047
  int k   = idx & 511;
  int g   = row >> 9;                             // 0..3 : f,i,g,o
  int rr  = row & 511;
  const float* W = (g == 0) ? Wf : (g == 1) ? Wi : (g == 2) ? Wg : Wo;
  Wx[(size_t)row * 512 + k] = (bf16)W[(size_t)rr * (DIN + HID) + k];
  Wh[(size_t)row * 512 + k] = (bf16)W[(size_t)rr * (DIN + HID) + DIN + k];
}

// ---------------- convert inputs fp32 -> bf16 ----------------
__global__ void k_f2b(const float* __restrict__ src, bf16* __restrict__ dst, size_t n) {
  size_t i = (size_t)blockIdx.x * 256 + threadIdx.x;
  size_t stride = (size_t)gridDim.x * 256;
  for (; i < n; i += stride) dst[i] = (bf16)src[i];
}

// ---------------- phase 1: xp = inputs @ Wx^T + b, stored in C-fragment swizzle ----------------
// One wave computes one 16x16 tile. Tiles: 4096 (M) x 128 (N).
__global__ void __launch_bounds__(256) k_gemm1(
    const bf16* __restrict__ A, const bf16* __restrict__ Wx,
    const float* __restrict__ bf_, const float* __restrict__ bi_,
    const float* __restrict__ bg_, const float* __restrict__ bo_,
    bf16* __restrict__ xp) {
  int wave = (blockIdx.x * blockDim.x + threadIdx.x) >> 5;
  int lane = threadIdx.x & 31;
  int r = lane & 15, half = lane >> 4;
  int mtile = wave >> 7;       // 0..4095  (row block of T*B)
  int ntile = wave & 127;      // 0..127   (col block of 4H)
  v8f acc = {};
  #pragma unroll 4
  for (int k0 = 0; k0 < DIN; k0 += 32) {
    v16bf a = load_frag(A,  mtile * 16, k0, DIN, r, half);
    v16bf b = load_frag(Wx, ntile * 16, k0, DIN, r, half);
    acc = __builtin_amdgcn_wmma_f32_16x16x32_bf16(false, a, false, b, (short)0, acc, false, false);
  }
  int col = ntile * 16 + r;    // global gate column (uniform bias per lane across slots)
  float bias = (col < 512)  ? bf_[col]
             : (col < 1024) ? bi_[col - 512]
             : (col < 1536) ? bg_[col - 1024]
                            : bo_[col - 1536];
  Pack8 pk;
  #pragma unroll
  for (int v = 0; v < 8; ++v) pk.h[v] = (bf16)(acc[v] + bias);
  // fragment-swizzled store: [tile][lane][slot], 16B contiguous per lane
  *(uint4*)(xp + (((size_t)wave) * 32 + lane) * 8) = pk.u;
}

// ---------------- phase 2: persistent LSTM scan ----------------
// 16 blocks x 256 threads. Block `blk` owns hidden cols [blk*32, blk*32+32).
// Wave (mtile = w&3, jsub = w>>2) owns batch rows mtile*16.. and cols jsub*16..,
// and computes ALL FOUR gate tiles for that patch -> cell update is wave-local.
__global__ void __launch_bounds__(256, 1) k_scan(
    const bf16* __restrict__ xp, const bf16* __restrict__ Wh,
    bf16* __restrict__ hbuf0, bf16* __restrict__ hbuf1,
    float* __restrict__ out, unsigned* __restrict__ barcnt) {
  __shared__ bf16 Hs[BATCH * LDH];

  const int tid  = threadIdx.x;
  const int blk  = blockIdx.x;          // 0..15
  const int wave = tid >> 5;
  const int lane = tid & 31;
  const int r = lane & 15, half = lane >> 4;
  const int mtile = wave & 3;           // batch tile
  const int jsub  = wave >> 2;          // 0..1 within SLICE

  const int jbase = blk * SLICE + jsub * 16;     // hidden col base for this wave
  float* hx_out = out + (size_t)T_STEPS * BATCH * HID;
  float* cx_out = hx_out + (size_t)BATCH * HID;

  v8f cstate = {};                      // cell state lives in registers for the whole scan

  for (int t = 0; t < T_STEPS; ++t) {
    const bf16* hin  = (t & 1) ? hbuf1 : hbuf0;
    bf16*       hout = (t & 1) ? hbuf0 : hbuf1;

    // stage h_{t-1} (64x512 bf16) into padded LDS via CDNA5 async copy
    // (ASYNCcnt-tracked, no VGPR round-trip): each lane moves 16B per op.
    {
      unsigned long long sbase = (unsigned long long)(size_t)hin;
      for (int idx = tid; idx < BATCH * (HID / 8); idx += 256) {
        int row = idx >> 6;             // 0..63
        int c8  = idx & 63;             // 8-elem chunk of 8 bf16
        unsigned voff    = (unsigned)(((unsigned)row * HID + c8 * 8) * 2);   // global byte offset
        unsigned lds_off = (unsigned)(size_t)(&Hs[row * LDH + c8 * 8]);      // LDS byte offset (low 32b)
        asm volatile("global_load_async_to_lds_b128 %0, %1, %2"
                     :: "v"(lds_off), "v"(voff), "s"(sbase) : "memory");
      }
      asm volatile("s_wait_asynccnt 0x0" ::: "memory");
    }
    __syncthreads();

    // warm L2/near caches for this step's x_proj fragments of the NEXT step
    {
      int tn = (t + 1 < T_STEPS) ? (t + 1) : t;
      size_t mt = (size_t)(tn * 4 + mtile) * 128;
      int nt = blk * 2 + jsub;
      __builtin_prefetch(xp + ((mt + (0 * 32 + nt)) * 32 + lane) * 8, 0, 1);
      __builtin_prefetch(xp + ((mt + (3 * 32 + nt)) * 32 + lane) * 8, 0, 1);
    }

    // z_patch[g] = h @ Wh[g-slice]^T  (accumulate fp32 via WMMA bf16)
    v8f acc0 = {}, acc1 = {}, acc2 = {}, acc3 = {};
    for (int k0 = 0; k0 < HID; k0 += 32) {
      const bf16* ap = &Hs[(mtile * 16 + r) * LDH + k0 + half * 8];
      FragBF fa;
      fa.u[0] = *(const uint4*)(ap);
      fa.u[1] = *(const uint4*)(ap + 16);
      v16bf b0 = load_frag(Wh, 0 * HID + jbase, k0, HID, r, half);
      v16bf b1 = load_frag(Wh, 1 * HID + jbase, k0, HID, r, half);
      v16bf b2 = load_frag(Wh, 2 * HID + jbase, k0, HID, r, half);
      v16bf b3 = load_frag(Wh, 3 * HID + jbase, k0, HID, r, half);
      acc0 = __builtin_amdgcn_wmma_f32_16x16x32_bf16(false, fa.v, false, b0, (short)0, acc0, false, false);
      acc1 = __builtin_amdgcn_wmma_f32_16x16x32_bf16(false, fa.v, false, b1, (short)0, acc1, false, false);
      acc2 = __builtin_amdgcn_wmma_f32_16x16x32_bf16(false, fa.v, false, b2, (short)0, acc2, false, false);
      acc3 = __builtin_amdgcn_wmma_f32_16x16x32_bf16(false, fa.v, false, b3, (short)0, acc3, false, false);
    }

    // fetch the 4 x_proj fragments (fragment-swizzled, one 16B load each)
    Pack8 xf, xi, xg, xo;
    {
      size_t mt = (size_t)(t * 4 + mtile) * 128;
      int nt = blk * 2 + jsub;
      xf.u = *(const uint4*)(xp + ((mt + (0 * 32 + nt)) * 32 + lane) * 8);
      xi.u = *(const uint4*)(xp + ((mt + (1 * 32 + nt)) * 32 + lane) * 8);
      xg.u = *(const uint4*)(xp + ((mt + (2 * 32 + nt)) * 32 + lane) * 8);
      xo.u = *(const uint4*)(xp + ((mt + (3 * 32 + nt)) * 32 + lane) * 8);
    }

    // gates + cell update, write h_t
    #pragma unroll
    for (int v = 0; v < 8; ++v) {
      float f = sigm_f(acc0[v] + (float)xf.h[v]);
      float i = sigm_f(acc1[v] + (float)xi.h[v]);
      float g = tanh_f(acc2[v] + (float)xg.h[v]);
      float o = sigm_f(acc3[v] + (float)xo.h[v]);
      float c = f * cstate[v] + i * g;
      cstate[v] = c;
      float h = o * tanh_f(c);
      int batch = mtile * 16 + v + 8 * half;     // C-fragment row mapping
      int j     = jbase + r;                     // C-fragment col mapping
      out[((size_t)t * BATCH + batch) * HID + j] = h;
      hout[(size_t)batch * HID + j] = (bf16)h;
      if (t == T_STEPS - 1) {
        hx_out[(size_t)batch * HID + j] = h;
        cx_out[(size_t)batch * HID + j] = c;
      }
    }

    // one grid-wide barrier per step (double-buffered h makes a single barrier safe)
    __threadfence();
    __syncthreads();
    if (tid == 0) {
      atomicAdd(barcnt, 1u);
      unsigned target = (unsigned)(t + 1) * NBLK;
      while (__hip_atomic_load(barcnt, __ATOMIC_ACQUIRE, __HIP_MEMORY_SCOPE_AGENT) < target)
        __builtin_amdgcn_s_sleep(2);
    }
    __syncthreads();
  }
}

extern "C" void kernel_launch(void* const* d_in, const int* in_sizes, int n_in,
                              void* d_out, int out_size, void* d_ws, size_t ws_size,
                              hipStream_t stream) {
  const float* inputs = (const float*)d_in[0];
  const float* Wf = (const float*)d_in[1];
  const float* bf_ = (const float*)d_in[2];
  const float* Wi = (const float*)d_in[3];
  const float* bi_ = (const float*)d_in[4];
  const float* Wg = (const float*)d_in[5];
  const float* bg_ = (const float*)d_in[6];
  const float* Wo = (const float*)d_in[7];
  const float* bo_ = (const float*)d_in[8];
  float* out = (float*)d_out;

  char* ws = (char*)d_ws;
  size_t off = 0;
  unsigned* cnt = (unsigned*)(ws + off); off += 256;
  bf16* h0  = (bf16*)(ws + off); off += (size_t)BATCH * HID * 2;
  bf16* h1  = (bf16*)(ws + off); off += (size_t)BATCH * HID * 2;
  bf16* Wx  = (bf16*)(ws + off); off += (size_t)G4 * DIN * 2;
  bf16* Wh  = (bf16*)(ws + off); off += (size_t)G4 * HID * 2;
  bf16* inb = (bf16*)(ws + off); off += (size_t)TB * DIN * 2;
  bf16* xp  = (bf16*)(ws + off); off += (size_t)TB * G4 * 2;   // fragment-swizzled x_proj

  k_init <<<128, 256, 0, stream>>>(cnt, h0, h1);
  k_cvtW <<<(G4 * 512) / 256, 256, 0, stream>>>(Wf, Wi, Wg, Wo, Wx, Wh);
  k_f2b  <<<32768, 256, 0, stream>>>(inputs, inb, (size_t)TB * DIN);
  k_gemm1<<<65536, 256, 0, stream>>>(inb, Wx, bf_, bi_, bg_, bo_, xp);
  k_scan <<<NBLK, 256, 0, stream>>>(xp, Wh, h0, h1, out, cnt);
}
